// FrameAttention_19670950216584
// MI455X (gfx1250) — matrix operand order repaired and checked by reading, not compiled, not run on hardware
//
#include <hip/hip_runtime.h>

// ---------------------------------------------------------------------------
// FrameAttention for MI455X (gfx1250, wave32, WMMA), register-blocked rev.
//   B=2, Cin=8, H=W=16, T=512, CH=64 heads, d=H*W=256
// ---------------------------------------------------------------------------

typedef _Float16 half8  __attribute__((ext_vector_type(8)));
typedef _Float16 half16 __attribute__((ext_vector_type(16)));
typedef float    v8f    __attribute__((ext_vector_type(8)));
typedef float    v4f    __attribute__((ext_vector_type(4)));
typedef float    v2f    __attribute__((ext_vector_type(2)));

#define NB   2
#define CIN  8
#define CHH  64
#define HH   16
#define WW   16
#define TT   512
#define DD   256                       // H*W
#define NTT  (TT * TT)                 // 262144 per (b,h)

union HFrag { half16 v; half8 h[2]; };

// A fragment (16x32 f16) at (rowBase, kk):  lane m<16 -> row m, K {0..7,16..23};
// lane m+16 -> row m, K {8..15,24..31}.  rowPtr already includes lane row.
__device__ __forceinline__ half16 load_a(const _Float16* rowPtr, int kk, int hi) {
    HFrag a;
    a.h[0] = *(const half8*)(rowPtr + kk + hi * 8);
    a.h[1] = *(const half8*)(rowPtr + kk + hi * 8 + 16);
    return a.v;
}
// B fragment (32x16 f16): lane n<16 -> col n, K {0..7,16..23}; lane n+16 -> K {8..15,24..31}.
__device__ __forceinline__ half16 load_b(const _Float16* colPtr, int kk, int hi) {
    HFrag b;
    b.h[0] = *(const half8*)(colPtr + kk + hi * 8);
    b.h[1] = *(const half8*)(colPtr + kk + hi * 8 + 16);
    return b.v;
}

// ---------------------------------------------------------------------------
// K1: fused q/k/v 3x3 spatial conv, f32 in -> f16 out.
//   qh, kh : (b,h, t, d) row-major ;  vt : (b,h, d, t) transposed
// ---------------------------------------------------------------------------
__global__ __launch_bounds__(512)
void k_conv_qkv(const float* __restrict__ x,
                const float* __restrict__ wq,
                const float* __restrict__ wk,
                const float* __restrict__ wv,
                _Float16* __restrict__ qh,
                _Float16* __restrict__ kh,
                _Float16* __restrict__ vt)
{
    __shared__ float swq[72], swk[72], swv[72];   // 8*3*3 per head
    const int ch = blockIdx.y;
    const int b  = blockIdx.z;
    const int yx = blockIdx.x;                    // 0..255 == d
    const int y  = yx >> 4, xw = yx & 15;
    const int t  = threadIdx.x;                   // 0..511

    if (threadIdx.x < 72) {
        swq[threadIdx.x] = wq[ch * 72 + threadIdx.x];
        swk[threadIdx.x] = wk[ch * 72 + threadIdx.x];
        swv[threadIdx.x] = wv[ch * 72 + threadIdx.x];
    }
    __syncthreads();

    float sq = 0.f, sk = 0.f, sv = 0.f;
    #pragma unroll
    for (int ci = 0; ci < CIN; ci++) {
        #pragma unroll
        for (int ky = 0; ky < 3; ky++) {
            const int yy = y + ky - 1;
            if ((unsigned)yy >= 16u) continue;
            #pragma unroll
            for (int kx = 0; kx < 3; kx++) {
                const int xx = xw + kx - 1;
                if ((unsigned)xx >= 16u) continue;
                const float xin = x[((((size_t)b * CIN + ci) * HH + yy) * WW + xx) * TT + t];
                const int wi = (ci * 3 + ky) * 3 + kx;
                sq = fmaf(xin, swq[wi], sq);
                sk = fmaf(xin, swk[wi], sk);
                sv = fmaf(xin, swv[wi], sv);
            }
        }
    }
    const size_t bh = (size_t)b * CHH + ch;
    qh[(bh * TT + t) * DD + yx] = (_Float16)sq;
    kh[(bh * TT + t) * DD + yx] = (_Float16)sk;
    vt[(bh * DD + yx) * TT + t] = (_Float16)sv;   // coalesced over t
}

// ---------------------------------------------------------------------------
// K2: scores = mask( (q @ k^T + alibi) / 16 ).  f16 WMMA, f32 out.
//   Register-blocked: each wave -> 32(i) x 64(j) = 8 accumulators.
//   Block (8 waves) covers 32(i) x 512(all j).  Grid: bh(128) x iblk(16).
// ---------------------------------------------------------------------------
__global__ __launch_bounds__(256)
void k_scores(const _Float16* __restrict__ qh,
              const _Float16* __restrict__ kh,
              float* __restrict__ scores)
{
    const int bx    = blockIdx.x;
    const int iblk  = bx & 15;
    const int bh    = bx >> 4;
    const int h     = bh & (CHH - 1);
    const int lane  = threadIdx.x & 31;
    const int wave  = threadIdx.x >> 5;
    const int ibase = iblk * 32;
    const int jbase = wave * 64;
    const int row   = lane & 15;
    const int hi    = lane >> 4;

    const _Float16* arow0 = qh + ((size_t)bh * TT + ibase + row) * DD;
    const _Float16* arow1 = arow0 + (size_t)16 * DD;
    const _Float16* brow0 = kh + ((size_t)bh * TT + jbase + row) * DD;

    v8f acc[2][4] = {};
    #pragma unroll
    for (int kk = 0; kk < DD; kk += 32) {
        const half16 a0 = load_a(arow0, kk, hi);
        const half16 a1 = load_a(arow1, kk, hi);
        if (kk + 32 < DD) {
            __builtin_prefetch(arow0 + kk + 32, 0, 0);
            __builtin_prefetch(brow0 + kk + 32, 0, 0);
        }
        #pragma unroll
        for (int jt = 0; jt < 4; jt++) {
            const half16 b = load_b(brow0 + (size_t)jt * 16 * DD, kk, hi);
            acc[0][jt] = __builtin_amdgcn_wmma_f32_16x16x32_f16(false, a0, false, b,
                                                                (short)0, acc[0][jt], false, false);
            acc[1][jt] = __builtin_amdgcn_wmma_f32_16x16x32_f16(false, a1, false, b,
                                                                (short)0, acc[1][jt], false, false);
        }
    }

    // epilogue: ALiBi (slope only on first 8 heads), scale, lower-tri mask
    const float slope = (h < 8) ? exp2f(-(float)(h + 1)) : 0.0f;
    #pragma unroll
    for (int ih = 0; ih < 2; ih++) {
        #pragma unroll
        for (int jt = 0; jt < 4; jt++) {
            const int j = jbase + jt * 16 + (lane & 15);
            #pragma unroll
            for (int r = 0; r < 8; r++) {
                const int i = ibase + ih * 16 + r + hi * 8;   // C/D: M = r + 8*hi
                float val = acc[ih][jt][r] - slope * fabsf((float)(i - j));
                val *= 0.0625f;                               // 1/sqrt(256)
                if (j <= i) val = -1e18f;                     // tril incl. diag
                scores[((size_t)bh * TT + i) * TT + j] = val;
            }
        }
    }
}

// ---------------------------------------------------------------------------
// K3/K5: talking-heads mix.  out[b,g,n] = sum_h W[g,h] * S[b,h,n], n=(i,j).
//   Exact f32 math via V_WMMA_F32_16X16X4_F32 (masked -1e18 won't fit in f16).
//   Each wave computes ALL 64 g for its 16-column strip -> S read exactly once.
// A frag (16x4 f32): lane m<16: {K0,K1}; lane m+16: {K2,K3}.
// B frag (4x16 f32): lane n<16: {K0,K1}; lane n+16: {K2,K3} (col = lane&15).
// ---------------------------------------------------------------------------
template <bool HALF_OUT>
__global__ __launch_bounds__(256)
void k_mix(const float* __restrict__ S,
           const float* __restrict__ Wm,       // (CH, CH) row-major [g,h]
           float* __restrict__ outF,
           _Float16* __restrict__ outH)
{
    __shared__ float w[CHH * CHH];              // 16 KB
    for (int i = threadIdx.x; i < CHH * CHH; i += 256) w[i] = Wm[i];
    __syncthreads();

    const int b     = blockIdx.y;
    const int lane  = threadIdx.x & 31;
    const int wave  = threadIdx.x >> 5;
    const int nbase = blockIdx.x * 128 + wave * 16;
    const int m     = lane & 15;
    const int ks    = (lane >> 4) * 2;

    const float* Sb = S + (size_t)b * CHH * NTT;

    v8f acc[4] = {};
    #pragma unroll 4
    for (int hh = 0; hh < CHH; hh += 4) {
        v2f bm;
        bm.x = Sb[(size_t)(hh + ks)     * NTT + nbase + m];
        bm.y = Sb[(size_t)(hh + ks + 1) * NTT + nbase + m];
        #pragma unroll
        for (int gt = 0; gt < 4; gt++) {
            v2f a;
            a.x = w[(gt * 16 + m) * CHH + hh + ks];
            a.y = w[(gt * 16 + m) * CHH + hh + ks + 1];
            acc[gt] = __builtin_amdgcn_wmma_f32_16x16x4_f32(false, a, false, bm,
                                                            (short)0, acc[gt], false, false);
        }
    }
    #pragma unroll
    for (int gt = 0; gt < 4; gt++) {
        #pragma unroll
        for (int r = 0; r < 8; r++) {
            const int g = gt * 16 + r + (lane >> 4) * 8;
            const size_t oi = ((size_t)b * CHH + g) * NTT + nbase + m;
            if (HALF_OUT) outH[oi] = (_Float16)acc[gt][r];
            else          outF[oi] = acc[gt][r];
        }
    }
}

// ---------------------------------------------------------------------------
// K4: row softmax over j (512 elems).  One wave32 per row, 16 elems/lane.
// ---------------------------------------------------------------------------
__global__ __launch_bounds__(256)
void k_softmax(const float* __restrict__ mixed, float* __restrict__ p)
{
    const int lane = threadIdx.x & 31;
    const int wave = threadIdx.x >> 5;
    const size_t rowi = (size_t)blockIdx.x * 8 + wave;   // 65536 rows
    const float* src = mixed + rowi * TT;
    float*       dst = p     + rowi * TT;

    v4f vals[4];
    float m = -3.4e38f;
    #pragma unroll
    for (int c = 0; c < 4; c++) {
        vals[c] = *(const v4f*)(src + c * 128 + lane * 4);
        #pragma unroll
        for (int e = 0; e < 4; e++) m = fmaxf(m, vals[c][e]);
    }
    #pragma unroll
    for (int off = 16; off > 0; off >>= 1) m = fmaxf(m, __shfl_xor(m, off, 32));

    float s = 0.f;
    #pragma unroll
    for (int c = 0; c < 4; c++)
        #pragma unroll
        for (int e = 0; e < 4; e++) { vals[c][e] = __expf(vals[c][e] - m); s += vals[c][e]; }
    #pragma unroll
    for (int off = 16; off > 0; off >>= 1) s += __shfl_xor(s, off, 32);

    const float inv = 1.0f / s;
    #pragma unroll
    for (int c = 0; c < 4; c++) {
        v4f o = vals[c] * inv;
        *(v4f*)(dst + c * 128 + lane * 4) = o;
    }
}

// ---------------------------------------------------------------------------
// K6: out[b,h,d,i] = sum_j p2[b,h,i,j] * v[b,h,j,d].  f16 WMMA.
//   Register-blocked: each wave -> 32(i) x 32(d) = 4 accumulators.
//   Block covers 32(i) x 256(all d).  Grid: bh(128) x iblk(16).
// ---------------------------------------------------------------------------
__global__ __launch_bounds__(256)
void k_out_gemm(const _Float16* __restrict__ p2,
                const _Float16* __restrict__ vt,
                _Float16* __restrict__ oh)
{
    const int bx    = blockIdx.x;
    const int iblk  = bx & 15;
    const int bh    = bx >> 4;
    const int lane  = threadIdx.x & 31;
    const int wave  = threadIdx.x >> 5;
    const int ibase = iblk * 32;
    const int dbase = wave * 32;
    const int row   = lane & 15;
    const int hi    = lane >> 4;

    const _Float16* arow0 = p2 + ((size_t)bh * TT + ibase + row) * TT;
    const _Float16* arow1 = arow0 + (size_t)16 * TT;
    const _Float16* brow0 = vt + ((size_t)bh * DD + dbase + row) * TT;
    const _Float16* brow1 = brow0 + (size_t)16 * TT;

    v8f acc[2][2] = {};
    #pragma unroll 4
    for (int jj = 0; jj < TT; jj += 32) {
        const half16 a0 = load_a(arow0, jj, hi);
        const half16 a1 = load_a(arow1, jj, hi);
        const half16 b0 = load_b(brow0, jj, hi);
        const half16 b1 = load_b(brow1, jj, hi);
        if (jj + 32 < TT) {
            __builtin_prefetch(arow0 + jj + 32, 0, 0);
            __builtin_prefetch(brow0 + jj + 32, 0, 0);
        }
        acc[0][0] = __builtin_amdgcn_wmma_f32_16x16x32_f16(false, a0, false, b0,
                                                           (short)0, acc[0][0], false, false);
        acc[0][1] = __builtin_amdgcn_wmma_f32_16x16x32_f16(false, a0, false, b1,
                                                           (short)0, acc[0][1], false, false);
        acc[1][0] = __builtin_amdgcn_wmma_f32_16x16x32_f16(false, a1, false, b0,
                                                           (short)0, acc[1][0], false, false);
        acc[1][1] = __builtin_amdgcn_wmma_f32_16x16x32_f16(false, a1, false, b1,
                                                           (short)0, acc[1][1], false, false);
    }
    #pragma unroll
    for (int ih = 0; ih < 2; ih++) {
        #pragma unroll
        for (int dt = 0; dt < 2; dt++) {
            const int d = dbase + dt * 16 + (lane & 15);
            #pragma unroll
            for (int r = 0; r < 8; r++) {
                const int i = ibase + ih * 16 + r + hi * 8;
                oh[((size_t)bh * DD + d) * TT + i] = (_Float16)acc[ih][dt][r];
            }
        }
    }
}

// ---------------------------------------------------------------------------
// K7: final 3x3 conv (64 -> 8 channels) + bias, f32 out.
// ---------------------------------------------------------------------------
__global__ __launch_bounds__(512)
void k_conv_out(const _Float16* __restrict__ oh,   // (b,h, d, t)
                const float* __restrict__ wp,      // (C, CH, 3,3)
                const float* __restrict__ bp,
                float* __restrict__ out)
{
    __shared__ float w[CHH * 9];                   // 2304 B
    const int co = blockIdx.z & 7;
    const int b  = blockIdx.z >> 3;
    const int y  = blockIdx.y;
    const int xw = blockIdx.x;
    const int t  = threadIdx.x;

    for (int i = threadIdx.x; i < CHH * 9; i += 512) w[i] = wp[co * CHH * 9 + i];
    __syncthreads();

    float s = bp[co];
    for (int ch = 0; ch < CHH; ch++) {
        const _Float16* base = oh + ((size_t)(b * CHH + ch) * DD) * TT;
        #pragma unroll
        for (int ky = 0; ky < 3; ky++) {
            const int yy = y + ky - 1;
            if ((unsigned)yy >= 16u) continue;
            #pragma unroll
            for (int kx = 0; kx < 3; kx++) {
                const int xx = xw + kx - 1;
                if ((unsigned)xx >= 16u) continue;
                s = fmaf((float)base[(size_t)(yy * 16 + xx) * TT + t],
                         w[(ch * 3 + ky) * 3 + kx], s);
            }
        }
    }
    out[((((size_t)b * CIN + co) * HH + y) * WW + xw) * TT + t] = s;
}

// ---------------------------------------------------------------------------
// Launch: x, wq, wk, wv, wp, bp, w_bs, w_as  (all f32)
// ---------------------------------------------------------------------------
extern "C" void kernel_launch(void* const* d_in, const int* in_sizes, int n_in,
                              void* d_out, int out_size, void* d_ws, size_t ws_size,
                              hipStream_t stream)
{
    const float* x    = (const float*)d_in[0];
    const float* wq   = (const float*)d_in[1];
    const float* wk   = (const float*)d_in[2];
    const float* wv   = (const float*)d_in[3];
    const float* wp   = (const float*)d_in[4];
    const float* bp   = (const float*)d_in[5];
    const float* w_bs = (const float*)d_in[6];
    const float* w_as = (const float*)d_in[7];
    float* out = (float*)d_out;

    // workspace layout (bytes)
    const size_t QKV_E  = (size_t)NB * CHH * TT * DD;        // 16.78M elems
    const size_t ATT_E  = (size_t)NB * CHH * TT * TT;        // 33.55M elems
    char* ws = (char*)d_ws;
    _Float16* qh     = (_Float16*)(ws);
    _Float16* kh     = (_Float16*)(ws + QKV_E * 2);
    _Float16* vt     = (_Float16*)(ws + QKV_E * 4);
    float*    scores = (float*)   (ws + QKV_E * 6);          // also reused for p
    float*    mixed  = (float*)   (ws + QKV_E * 6 + ATT_E * 4);
    _Float16* p2h    = (_Float16*)(ws + QKV_E * 6 + ATT_E * 8);
    _Float16* oh     = (_Float16*)(ws + QKV_E * 6 + ATT_E * 8 + ATT_E * 2);

    // 1) q/k/v convs -> f16
    k_conv_qkv<<<dim3(HH * WW, CHH, NB), 512, 0, stream>>>(x, wq, wk, wv, qh, kh, vt);

    // 2) scores = mask((q k^T + alibi)/16)   [f16 WMMA, f32 out]
    k_scores<<<dim3(NB * CHH * (TT / 32)), 256, 0, stream>>>(qh, kh, scores);

    // 3) talking-heads pre-softmax (f32 WMMA): mixed = w_bs @ scores
    k_mix<false><<<dim3(NTT / 128, NB), 256, 0, stream>>>(scores, w_bs, mixed, nullptr);

    // 4) softmax over j: p (f32) written over the scores buffer
    k_softmax<<<dim3((size_t)NB * CHH * TT / 8), 256, 0, stream>>>(mixed, scores);

    // 5) talking-heads post-softmax (f32 WMMA) -> f16 p2
    k_mix<true><<<dim3(NTT / 128, NB), 256, 0, stream>>>(scores, w_as, nullptr, p2h);

    // 6) out = p2 @ v   [f16 WMMA], 32x32 per wave, stored (d,t)-transposed
    k_out_gemm<<<dim3(NB * CHH * (TT / 32)), 256, 0, stream>>>(p2h, vt, oh);

    // 7) final conv + bias -> f32 output
    k_conv_out<<<dim3(WW, HH, NB * CIN), 512, 0, stream>>>(oh, wp, bp, out);
}
